// MoiraiModule_489626272172
// MI455X (gfx1250) — matrix operand order repaired and checked by reading, not compiled, not run on hardware
//
#include <hip/hip_runtime.h>
#include <hip/hip_bf16.h>
#include <stdint.h>

// ---------------------------------------------------------------------------
// Moirai-style transformer forward for MI455X (gfx1250), bf16 WMMA pipeline.
// B=4 S=512 D=768 L=4 H=12 HD=64 DFF=3072 MAXP=128, P_SIZES={32,64,128}
//
// GEMM: wave computes 32x64 (2 A-frags x 4 B-frags = 8 WMMA per 32-K step).
// K-loop unrolled x2 with ping-pong fragment buffers (no rotation moves) +
// global_prefetch one step ahead, so WMMAs overlap the other buffer's loads.
// ---------------------------------------------------------------------------

#define B_    4
#define S_    512
#define D_    768
#define L_    4
#define H_    12
#define HD_   64
#define DFF_  3072
#define MAXP_ 128
#define NTOK  (B_ * S_)
#define EPSF  1e-5f

typedef __bf16 bf16_t;
typedef __attribute__((ext_vector_type(16))) __bf16 v16bf;
typedef __attribute__((ext_vector_type(8)))  float  v8f;

union FragU { v16bf v; uint4 q[2]; };

// ---- WMMA helpers ----------------------------------------------------------

__device__ __forceinline__ v8f wmma_bf(v16bf a, v16bf b, v8f c) {
  return __builtin_amdgcn_wmma_f32_16x16x32_bf16(
      /*neg_a=*/false, a, /*neg_b=*/false, b,
      /*c_mod=*/(short)0, c, /*reuse_a=*/false, /*reuse_b=*/false);
}

// A-matrix fragment (16x32 bf16, M-rows x K): lanes 0-15 hold rows 0-15 with
// K={kb..kb+7, kb+16..kb+23}; lanes 16-31 hold rows 0-15 with K={kb+8.., kb+24..}.
__device__ __forceinline__ v16bf load_frag_a(const bf16_t* A, int lda, int m0, int k0) {
  const int lane = threadIdx.x & 31;
  const int row  = m0 + (lane & 15);
  const int kb   = k0 + ((lane & 16) ? 8 : 0);
  const bf16_t* p = A + (long)row * lda + kb;
  FragU f;
  f.q[0] = *(const uint4*)(p);
  f.q[1] = *(const uint4*)(p + 16);
  return f.v;
}

// B-matrix fragment (32x16 bf16, K x N-cols) loaded from BT where BT is
// row-major [N][K] (contiguous K per output column). Lanes 0-15 hold column n,
// K={k0..k0+7, k0+16..k0+23}; lanes 16-31 hold column n, K={k0+8.., k0+24..}.
__device__ __forceinline__ v16bf load_frag_b(const bf16_t* BT, int ldb, int n0, int k0) {
  const int lane = threadIdx.x & 31;
  const int col  = n0 + (lane & 15);
  const int kb   = k0 + ((lane & 16) ? 8 : 0);
  const bf16_t* p = BT + (long)col * ldb + kb;
  FragU f;
  f.q[0] = *(const uint4*)(p);
  f.q[1] = *(const uint4*)(p + 16);
  return f.v;
}

// ---- wave reductions (wave32) ---------------------------------------------

__device__ __forceinline__ float wave_sum32(float v) {
#pragma unroll
  for (int m = 16; m > 0; m >>= 1) v += __shfl_xor(v, m, 32);
  return v;
}
__device__ __forceinline__ float row_sum16(float v) {
#pragma unroll
  for (int m = 8; m > 0; m >>= 1) v += __shfl_xor(v, m, 32);
  return v;
}
__device__ __forceinline__ float row_max16(float v) {
#pragma unroll
  for (int m = 8; m > 0; m >>= 1) v = fmaxf(v, __shfl_xor(v, m, 32));
  return v;
}

// ---------------------------------------------------------------------------
// Weight convert + transpose: W[K][N] fp32 -> WT[N][K] bf16
// ---------------------------------------------------------------------------
__global__ __launch_bounds__(256) void transpose_bf16(
    const float* __restrict__ W, bf16_t* __restrict__ WT, int K, int N) {
  int i = blockIdx.x * 256 + threadIdx.x;
  if (i >= K * N) return;
  int k = i % K, n = i / K;
  WT[i] = (bf16_t)W[(long)k * N + n];
}

// ---------------------------------------------------------------------------
// Generic bf16 WMMA GEMM: C[m,n] = sum_k A[m,k] * BT[n,k] (+ bias[n])
// Block = 256 threads = 8 waves; wave computes 32x64; block tile 256x64.
// mode: 0 = store fp32, 1 = accumulate fp32 (C += r), 2 = store bf16,
//       3 = GLU: store bf16( silu(aux[idx]) * r )
// C index = m*sm + n*sn (lets V be stored directly in [B,H,HD,S] layout).
// ---------------------------------------------------------------------------
__global__ __launch_bounds__(256) void gemm_wmma(
    const bf16_t* __restrict__ A, int lda,
    const bf16_t* __restrict__ BT, int ldb,
    const float* __restrict__ bias,
    void* __restrict__ C, const float* __restrict__ aux,
    long sm, long sn, int M, int N, int K, int mode) {
  const int lane = threadIdx.x & 31;
  const int wid  = threadIdx.x >> 5;
  const int m0   = blockIdx.y * 256 + wid * 32;
  const int n0   = blockIdx.x * 64;
  if (m0 >= M || n0 >= N) return;

  v8f acc[2][4];
#pragma unroll
  for (int u = 0; u < 2; ++u)
#pragma unroll
    for (int j = 0; j < 4; ++j) acc[u][j] = (v8f){};

  // ping-pong fragment buffers: buffer0 holds K-step k, buffer1 holds k+32.
  // Each buffer is reloaded in place (no register rotation moves).
  v16bf a0[2], b0[4], a1[2], b1[4];
  a0[0] = load_frag_a(A, lda, m0, 0);
  a0[1] = load_frag_a(A, lda, m0 + 16, 0);
#pragma unroll
  for (int j = 0; j < 4; ++j) b0[j] = load_frag_b(BT, ldb, n0 + j * 16, 0);

  for (int k = 0; k < K; k += 64) {
    const int k1 = k + 32, k2 = k + 64;
    if (k1 < K) {   // preload buffer1 for k+32 while buffer0 computes
      a1[0] = load_frag_a(A, lda, m0, k1);
      a1[1] = load_frag_a(A, lda, m0 + 16, k1);
#pragma unroll
      for (int j = 0; j < 4; ++j) b1[j] = load_frag_b(BT, ldb, n0 + j * 16, k1);
      if (k2 < K) {  // pull the K-step after next into cache (gfx1250 prefetch)
        __builtin_prefetch(A + (long)(m0 + (lane & 15)) * lda + k2, 0, 1);
        __builtin_prefetch(A + (long)(m0 + 16 + (lane & 15)) * lda + k2, 0, 1);
        __builtin_prefetch(BT + (long)(n0 + (lane & 31)) * ldb + k2, 0, 1);
        __builtin_prefetch(BT + (long)(n0 + 32 + (lane & 31)) * ldb + k2, 0, 1);
      }
    }
#pragma unroll
    for (int j = 0; j < 4; ++j) {
      acc[0][j] = wmma_bf(a0[0], b0[j], acc[0][j]);
      acc[1][j] = wmma_bf(a0[1], b0[j], acc[1][j]);
    }
    if (k1 < K) {
      if (k2 < K) {  // reload buffer0 for k+64 while buffer1 computes
        a0[0] = load_frag_a(A, lda, m0, k2);
        a0[1] = load_frag_a(A, lda, m0 + 16, k2);
#pragma unroll
        for (int j = 0; j < 4; ++j) b0[j] = load_frag_b(BT, ldb, n0 + j * 16, k2);
        if (k2 + 32 < K) {
          __builtin_prefetch(A + (long)(m0 + (lane & 15)) * lda + k2 + 32, 0, 1);
          __builtin_prefetch(A + (long)(m0 + 16 + (lane & 15)) * lda + k2 + 32, 0, 1);
          __builtin_prefetch(BT + (long)(n0 + (lane & 31)) * ldb + k2 + 32, 0, 1);
          __builtin_prefetch(BT + (long)(n0 + 32 + (lane & 31)) * ldb + k2 + 32, 0, 1);
        }
      }
#pragma unroll
      for (int j = 0; j < 4; ++j) {
        acc[0][j] = wmma_bf(a1[0], b1[j], acc[0][j]);
        acc[1][j] = wmma_bf(a1[1], b1[j], acc[1][j]);
      }
    }
  }

#pragma unroll
  for (int u = 0; u < 2; ++u) {
#pragma unroll
    for (int j = 0; j < 4; ++j) {
      const int n = n0 + j * 16 + (lane & 15);
      const float bv = bias ? bias[n] : 0.0f;
#pragma unroll
      for (int v = 0; v < 8; ++v) {
        const int m = m0 + u * 16 + v + ((lane & 16) ? 8 : 0);
        const long idx = (long)m * sm + (long)n * sn;
        float r = acc[u][j][v] + bv;
        if (mode == 0) {
          ((float*)C)[idx] = r;
        } else if (mode == 1) {
          ((float*)C)[idx] += r;
        } else if (mode == 2) {
          ((bf16_t*)C)[idx] = (bf16_t)r;
        } else {
          float g = aux[idx];
          float s = g / (1.0f + __expf(-g));   // silu
          ((bf16_t*)C)[idx] = (bf16_t)(s * r);
        }
      }
    }
  }
}

// ---------------------------------------------------------------------------
// Packed std scaler kernels
// ---------------------------------------------------------------------------
__global__ __launch_bounds__(256) void tok_stats(
    const float* __restrict__ tgt, const uint8_t* __restrict__ obs,
    const uint8_t* __restrict__ pred, float* __restrict__ cnt,
    float* __restrict__ s1) {
  const int t = blockIdx.x * 8 + (threadIdx.x >> 5);
  const int lane = threadIdx.x & 31;
  const int pm = pred[t];
  float c = 0.f, s = 0.f;
#pragma unroll
  for (int i = 0; i < 4; ++i) {
    const int f = lane + i * 32;
    const float m = (obs[(long)t * MAXP_ + f] && !pm) ? 1.f : 0.f;
    c += m;
    s += tgt[(long)t * MAXP_ + f] * m;
  }
  c = wave_sum32(c);
  s = wave_sum32(s);
  if (lane == 0) { cnt[t] = c; s1[t] = s; }
}

__global__ __launch_bounds__(256) void group_loc(
    const int* __restrict__ sid, const int* __restrict__ vid,
    const float* __restrict__ cnt, const float* __restrict__ s1,
    float* __restrict__ loc, float* __restrict__ tobs) {
  const int t = blockIdx.x * 256 + threadIdx.x;
  if (t >= NTOK) return;
  const int b = t / S_;
  const int ms = sid[t], mv = vid[t];
  float to = 0.f, ls = 0.f;
  for (int j = 0; j < S_; ++j) {
    const int jj = b * S_ + j;
    if (sid[jj] == ms && vid[jj] == mv) { to += cnt[jj]; ls += s1[jj]; }
  }
  loc[t]  = (to > 0.f) ? ls / fmaxf(to, 1.f) : 0.f;
  tobs[t] = to;
}

__global__ __launch_bounds__(256) void var_tok_k(
    const float* __restrict__ tgt, const uint8_t* __restrict__ obs,
    const uint8_t* __restrict__ pred, const float* __restrict__ loc,
    float* __restrict__ vtok) {
  const int t = blockIdx.x * 8 + (threadIdx.x >> 5);
  const int lane = threadIdx.x & 31;
  const int pm = pred[t];
  const float lo = loc[t];
  float acc = 0.f;
#pragma unroll
  for (int i = 0; i < 4; ++i) {
    const int f = lane + i * 32;
    const float m = (obs[(long)t * MAXP_ + f] && !pm) ? 1.f : 0.f;
    const float d = tgt[(long)t * MAXP_ + f] - lo;
    acc += d * d * m;
  }
  acc = wave_sum32(acc);
  if (lane == 0) vtok[t] = acc;
}

__global__ __launch_bounds__(256) void group_scale(
    const int* __restrict__ sid, const int* __restrict__ vid,
    const float* __restrict__ vtok, const float* __restrict__ tobs,
    float* __restrict__ scalev) {
  const int t = blockIdx.x * 256 + threadIdx.x;
  if (t >= NTOK) return;
  const int b = t / S_;
  const int ms = sid[t], mv = vid[t];
  float vs = 0.f;
  for (int j = 0; j < S_; ++j) {
    const int jj = b * S_ + j;
    if (sid[jj] == ms && vid[jj] == mv) vs += vtok[jj];
  }
  const float to = tobs[t];
  const float var = vs / fmaxf(to - 1.f, 1.f);
  scalev[t] = (to > 0.f) ? sqrtf(var + EPSF) : 1.f;
}

__global__ __launch_bounds__(256) void scale_input(
    const float* __restrict__ tgt, const float* __restrict__ loc,
    const float* __restrict__ scalev, bf16_t* __restrict__ out) {
  const int i = blockIdx.x * 256 + threadIdx.x;
  if (i >= NTOK * MAXP_) return;
  const int t = i >> 7;
  out[i] = (bf16_t)((tgt[i] - loc[t]) / scalev[t]);
}

// ---------------------------------------------------------------------------
// Select patch-size projection + mask-embedding replacement
// ---------------------------------------------------------------------------
__global__ __launch_bounds__(256) void combine_x(
    const float* __restrict__ xin, const int* __restrict__ psize,
    const uint8_t* __restrict__ pmask, const float* __restrict__ memb,
    float* __restrict__ x) {
  const int i = blockIdx.x * 256 + threadIdx.x;
  if (i >= NTOK * D_) return;
  const int t = i / D_, d = i % D_;
  const int ps = psize[t];
  const int p = (ps == 32) ? 0 : (ps == 64) ? 1 : 2;
  float v = xin[(long)p * NTOK * D_ + i];
  if (pmask[t]) v = memb[d];
  x[i] = v;
}

// ---------------------------------------------------------------------------
// RMSNorm over D=768, output bf16. One block per token.
// ---------------------------------------------------------------------------
__global__ __launch_bounds__(256) void rmsnorm_768(
    const float* __restrict__ x, const float* __restrict__ w,
    bf16_t* __restrict__ h) {
  __shared__ float wsum[8];
  const int t = blockIdx.x;
  const int tid = threadIdx.x;
  const long base = (long)t * D_;
  const float v0 = x[base + tid];
  const float v1 = x[base + tid + 256];
  const float v2 = x[base + tid + 512];
  float ss = wave_sum32(v0 * v0 + v1 * v1 + v2 * v2);
  if ((tid & 31) == 0) wsum[tid >> 5] = ss;
  __syncthreads();
  float tot = 0.f;
#pragma unroll
  for (int i = 0; i < 8; ++i) tot += wsum[i];
  const float rr = rsqrtf(tot / (float)D_ + EPSF);
  h[base + tid]       = (bf16_t)(v0 * rr * w[tid]);
  h[base + tid + 256] = (bf16_t)(v1 * rr * w[tid + 256]);
  h[base + tid + 512] = (bf16_t)(v2 * rr * w[tid + 512]);
}

// ---------------------------------------------------------------------------
// QK-RMSNorm (per head, 64 dims) + partial RoPE (first 32 dims), fp32 -> bf16.
// One wave per (token, head). Lane owns dims {lane, lane+32}.
// ---------------------------------------------------------------------------
__global__ __launch_bounds__(256) void qknorm_rope(
    const float* __restrict__ Xin, bf16_t* __restrict__ Xout,
    const float* __restrict__ w, const int* __restrict__ time_id) {
  const int lane = threadIdx.x & 31;
  const int g = blockIdx.x * 8 + (threadIdx.x >> 5);  // (b*S+s)*H + h
  const int h = g % H_;
  const int ts = g / H_;
  const long base = (long)ts * D_ + h * HD_;
  float x0 = Xin[base + lane];
  float x1 = Xin[base + lane + 32];
  const float ss = wave_sum32(x0 * x0 + x1 * x1);
  const float rr = rsqrtf(ss / (float)HD_ + EPSF);
  x0 *= rr * w[lane];
  x1 *= rr * w[lane + 32];
  // RoPE on dims [0,32): pairs (2j, 2j+1), inv_freq = 10000^{-j/16}
  const int tpos = time_id[ts];
  const int j = lane >> 1;
  const float inv = __expf(-(float)j * (9.210340372f / 16.0f)); // ln(10000)
  const float ang = (float)tpos * inv;
  const float c = __cosf(ang), s = __sinf(ang);
  const float partner = __shfl_xor(x0, 1, 32);
  const float y = x0 * c + partner * ((lane & 1) ? s : -s);
  Xout[base + lane]      = (bf16_t)y;
  Xout[base + lane + 32] = (bf16_t)x1;
}

// ---------------------------------------------------------------------------
// Fused flash attention. One wave per (b, h, 16-query tile); 32 keys / iter.
// Q,K in [B,S,H,HD] bf16 (ld=D); V in transposed [B,H,HD,S] bf16 (ld=S).
// Online softmax with -3e38 mask sentinel; P staged via LDS into an A-frag.
// ---------------------------------------------------------------------------
__global__ __launch_bounds__(256) void attn_fused(
    const bf16_t* __restrict__ Qb, const bf16_t* __restrict__ Kb,
    const bf16_t* __restrict__ VT, bf16_t* __restrict__ Ob,
    const int* __restrict__ sid, const int* __restrict__ vid,
    const float* __restrict__ vbias, int layer) {
  __shared__ bf16_t plds[8][16 * 32];
  const int lane = threadIdx.x & 31;
  const int wid  = threadIdx.x >> 5;
  const int g = blockIdx.x * 8 + wid;
  const int qt = g & 31;            // S/16 = 32 query tiles
  const int h  = (g >> 5) % H_;
  const int b  = g / (32 * H_);

  const long qkbase = (long)b * S_ * D_ + h * HD_;
  const bf16_t* Qp = Qb + qkbase;
  const bf16_t* Kp = Kb + qkbase;
  const bf16_t* Vp = VT + ((long)b * D_ + h * HD_) * S_;

  const v16bf qa0 = load_frag_a(Qp, D_, qt * 16, 0);
  const v16bf qa1 = load_frag_a(Qp, D_, qt * 16, 32);

  int qsam[8], qvar[8];
#pragma unroll
  for (int v = 0; v < 8; ++v) {
    const int q = qt * 16 + v + ((lane & 16) ? 8 : 0);
    qsam[v] = sid[b * S_ + q];
    qvar[v] = vid[b * S_ + q];
  }
  const float bias0 = vbias[(layer * 2 + 0) * H_ + h];
  const float bias1 = vbias[(layer * 2 + 1) * H_ + h];

  float mrow[8], lrow[8];
  v8f oacc[4];
#pragma unroll
  for (int v = 0; v < 8; ++v) { mrow[v] = -3e38f; lrow[v] = 0.f; }
#pragma unroll
  for (int f = 0; f < 4; ++f) oacc[f] = (v8f){};

  for (int kt = 0; kt < S_ / 32; ++kt) {
    const int key0 = kt * 32;
    if (kt + 1 < S_ / 32) {  // prefetch next key tile rows (K and V)
      __builtin_prefetch(Kp + (long)(key0 + 32 + (lane & 31)) * D_, 0, 1);
      __builtin_prefetch(Vp + (long)(lane & 31) * S_ + key0 + 32, 0, 1);
    }
    float st[2][8];
#pragma unroll
    for (int t2 = 0; t2 < 2; ++t2) {
      const int kb16 = key0 + t2 * 16;
      const v16bf b0 = load_frag_b(Kp, D_, kb16, 0);
      const v16bf b1 = load_frag_b(Kp, D_, kb16, 32);
      v8f sc = (v8f){};
      sc = wmma_bf(qa0, b0, sc);
      sc = wmma_bf(qa1, b1, sc);
      const int ksam = sid[b * S_ + kb16 + (lane & 15)];
      const int kvar = vid[b * S_ + kb16 + (lane & 15)];
#pragma unroll
      for (int v = 0; v < 8; ++v) {
        float s = sc[v] * 0.125f;  // 1/sqrt(64)
        s += (qvar[v] == kvar) ? bias1 : bias0;
        st[t2][v] = (qsam[v] == ksam) ? s : -3e38f;
      }
    }
    // online softmax per query row (16 lanes hold the 16 key columns)
#pragma unroll
    for (int v = 0; v < 8; ++v) {
      float rm = row_max16(fmaxf(st[0][v], st[1][v]));
      const float nm = fmaxf(mrow[v], rm);
      const float corr = __expf(mrow[v] - nm);
      mrow[v] = nm;
      const float p0 = __expf(st[0][v] - nm);
      const float p1 = __expf(st[1][v] - nm);
      lrow[v] = lrow[v] * corr + row_sum16(p0 + p1);
#pragma unroll
      for (int f = 0; f < 4; ++f) oacc[f][v] = oacc[f][v] * corr;
      const int r = v + ((lane & 16) ? 8 : 0);
      plds[wid][r * 32 + (lane & 15)]      = (bf16_t)p0;
      plds[wid][r * 32 + 16 + (lane & 15)] = (bf16_t)p1;
    }
    // P (16x32) @ V (32x64): A-frag from LDS, V B-frags contiguous in VT.
    const v16bf pa = load_frag_a(&plds[wid][0], 32, 0, 0);
#pragma unroll
    for (int f = 0; f < 4; ++f) {
      const v16bf vb = load_frag_b(Vp, S_, f * 16, key0);
      oacc[f] = wmma_bf(pa, vb, oacc[f]);
    }
  }

#pragma unroll
  for (int f = 0; f < 4; ++f) {
    const int hd = f * 16 + (lane & 15);
#pragma unroll
    for (int v = 0; v < 8; ++v) {
      const int q = qt * 16 + v + ((lane & 16) ? 8 : 0);
      const float o = oacc[f][v] / fmaxf(lrow[v], 1e-20f);
      Ob[((long)(b * S_) + q) * D_ + h * HD_ + hd] = (bf16_t)o;
    }
  }
}

// ---------------------------------------------------------------------------
// Final head: select patch projection, affine transform, softplus.
// ---------------------------------------------------------------------------
__global__ __launch_bounds__(256) void final_out(
    const float* __restrict__ outp, const int* __restrict__ psize,
    const float* __restrict__ loc, const float* __restrict__ scalev,
    float* __restrict__ out) {
  const int i = blockIdx.x * 256 + threadIdx.x;
  if (i >= NTOK * MAXP_) return;
  const int t = i >> 7, j = i & 127;
  const int ps = psize[t];
  const int p = (ps == 32) ? 0 : (ps == 64) ? 1 : 2;
  const bool valid = j < ps;
  const float ol = valid ? outp[(long)p * NTOK * MAXP_ + i] : 0.f;
  const float os = valid ? outp[(long)(3 + p) * NTOK * MAXP_ + i] : 0.f;
  const float sp = (os > 20.f) ? os : log1pf(__expf(os));
  out[i]                  = loc[t] + scalev[t] * ol;
  out[NTOK * MAXP_ + i]   = scalev[t] * sp;
}

// ---------------------------------------------------------------------------
// Host launcher
// ---------------------------------------------------------------------------
extern "C" void kernel_launch(void* const* d_in, const int* in_sizes, int n_in,
                              void* d_out, int out_size, void* d_ws, size_t ws_size,
                              hipStream_t stream) {
  (void)in_sizes; (void)n_in; (void)out_size; (void)ws_size;

  const float*   target     = (const float*)d_in[0];
  const uint8_t* observed   = (const uint8_t*)d_in[1];   // jnp bool_: 1 byte
  const int*     sample_id  = (const int*)d_in[2];
  const int*     time_id    = (const int*)d_in[3];
  const int*     variate_id = (const int*)d_in[4];
  const uint8_t* pred_mask  = (const uint8_t*)d_in[5];   // jnp bool_
  const int*     patch_sz   = (const int*)d_in[6];
  const float*   mask_emb   = (const float*)d_in[7];
  const float*   W_in       = (const float*)d_in[8];
  const float*   b_in       = (const float*)d_in[9];
  const float*   Wq         = (const float*)d_in[10];
  const float*   Wk         = (const float*)d_in[11];
  const float*   Wv         = (const float*)d_in[12];
  const float*   Wo         = (const float*)d_in[13];
  const float*   q_norm_w   = (const float*)d_in[14];
  const float*   k_norm_w   = (const float*)d_in[15];
  const float*   norm1_w    = (const float*)d_in[16];
  const float*   norm2_w    = (const float*)d_in[17];
  const float*   var_bias   = (const float*)d_in[18];
  const float*   W1         = (const float*)d_in[19];
  const float*   W2         = (const float*)d_in[20];
  const float*   W3         = (const float*)d_in[21];
  const float*   fnorm_w    = (const float*)d_in[22];
  const float*   W_loc      = (const float*)d_in[23];
  const float*   b_loc      = (const float*)d_in[24];
  const float*   W_scale    = (const float*)d_in[25];
  const float*   b_scale    = (const float*)d_in[26];

  // ---- workspace carve (256B-aligned) ----
  char* wsp = (char*)d_ws;
  size_t off = 0;
  auto carve = [&](size_t bytes) -> char* {
    char* p = wsp + off;
    off += (bytes + 255) & ~(size_t)255;
    return p;
  };
  bf16_t* WTin = (bf16_t*)carve((size_t)3 * MAXP_ * D_ * 2);
  bf16_t* WTq  = (bf16_t*)carve((size_t)L_ * D_ * D_ * 2);
  bf16_t* WTk  = (bf16_t*)carve((size_t)L_ * D_ * D_ * 2);
  bf16_t* WTv  = (bf16_t*)carve((size_t)L_ * D_ * D_ * 2);
  bf16_t* WTo  = (bf16_t*)carve((size_t)L_ * D_ * D_ * 2);
  bf16_t* WT1  = (bf16_t*)carve((size_t)L_ * D_ * DFF_ * 2);
  bf16_t* WT3  = (bf16_t*)carve((size_t)L_ * D_ * DFF_ * 2);
  bf16_t* WT2  = (bf16_t*)carve((size_t)L_ * DFF_ * D_ * 2);
  bf16_t* WTl  = (bf16_t*)carve((size_t)3 * D_ * MAXP_ * 2);
  bf16_t* WTs  = (bf16_t*)carve((size_t)3 * D_ * MAXP_ * 2);
  float*  cnt    = (float*)carve(NTOK * 4);
  float*  s1     = (float*)carve(NTOK * 4);
  float*  tobs   = (float*)carve(NTOK * 4);
  float*  locb   = (float*)carve(NTOK * 4);
  float*  vtok   = (float*)carve(NTOK * 4);
  float*  scalev = (float*)carve(NTOK * 4);
  bf16_t* scaledb = (bf16_t*)carve((size_t)NTOK * MAXP_ * 2);
  float*  xin  = (float*)carve((size_t)3 * NTOK * D_ * 4);
  float*  x    = (float*)carve((size_t)NTOK * D_ * 4);
  bf16_t* hb   = (bf16_t*)carve((size_t)NTOK * D_ * 2);
  float*  qf   = (float*)carve((size_t)NTOK * D_ * 4);
  float*  kf   = (float*)carve((size_t)NTOK * D_ * 4);
  bf16_t* qb2  = (bf16_t*)carve((size_t)NTOK * D_ * 2);
  bf16_t* kb2  = (bf16_t*)carve((size_t)NTOK * D_ * 2);
  bf16_t* vt   = (bf16_t*)carve((size_t)NTOK * D_ * 2);   // [B,H,HD,S]
  bf16_t* ob   = (bf16_t*)carve((size_t)NTOK * D_ * 2);
  float*  g1   = (float*)carve((size_t)NTOK * DFF_ * 4);
  bf16_t* gl   = (bf16_t*)carve((size_t)NTOK * DFF_ * 2);
  float*  outp = (float*)carve((size_t)6 * NTOK * MAXP_ * 4);

  // ---- weight transpose/convert to bf16 ----
  struct TJ { const float* s; bf16_t* d; int K, N; };
  TJ jobs[64]; int nj = 0;
  for (int p = 0; p < 3; ++p)
    jobs[nj++] = { W_in + (size_t)p * MAXP_ * D_, WTin + (size_t)p * D_ * MAXP_, MAXP_, D_ };
  for (int l = 0; l < L_; ++l) {
    jobs[nj++] = { Wq + (size_t)l * D_ * D_, WTq + (size_t)l * D_ * D_, D_, D_ };
    jobs[nj++] = { Wk + (size_t)l * D_ * D_, WTk + (size_t)l * D_ * D_, D_, D_ };
    jobs[nj++] = { Wv + (size_t)l * D_ * D_, WTv + (size_t)l * D_ * D_, D_, D_ };
    jobs[nj++] = { Wo + (size_t)l * D_ * D_, WTo + (size_t)l * D_ * D_, D_, D_ };
    jobs[nj++] = { W1 + (size_t)l * D_ * DFF_, WT1 + (size_t)l * DFF_ * D_, D_, DFF_ };
    jobs[nj++] = { W3 + (size_t)l * D_ * DFF_, WT3 + (size_t)l * DFF_ * D_, D_, DFF_ };
    jobs[nj++] = { W2 + (size_t)l * DFF_ * D_, WT2 + (size_t)l * D_ * DFF_, DFF_, D_ };
  }
  for (int p = 0; p < 3; ++p) {
    jobs[nj++] = { W_loc   + (size_t)p * D_ * MAXP_, WTl + (size_t)p * MAXP_ * D_, D_, MAXP_ };
    jobs[nj++] = { W_scale + (size_t)p * D_ * MAXP_, WTs + (size_t)p * MAXP_ * D_, D_, MAXP_ };
  }
  for (int i = 0; i < nj; ++i) {
    int tot = jobs[i].K * jobs[i].N;
    transpose_bf16<<<(tot + 255) / 256, 256, 0, stream>>>(jobs[i].s, jobs[i].d,
                                                          jobs[i].K, jobs[i].N);
  }

  // ---- packed std scaler ----
  tok_stats<<<NTOK / 8, 256, 0, stream>>>(target, observed, pred_mask, cnt, s1);
  group_loc<<<NTOK / 256, 256, 0, stream>>>(sample_id, variate_id, cnt, s1, locb, tobs);
  var_tok_k<<<NTOK / 8, 256, 0, stream>>>(target, observed, pred_mask, locb, vtok);
  group_scale<<<NTOK / 256, 256, 0, stream>>>(sample_id, variate_id, vtok, tobs, scalev);
  scale_input<<<NTOK * MAXP_ / 256, 256, 0, stream>>>(target, locb, scalev, scaledb);

  // ---- input projection (per patch size: masked cols == truncated K) ----
  const int psizes[3] = { 32, 64, 128 };
  for (int p = 0; p < 3; ++p) {
    gemm_wmma<<<dim3(D_ / 64, NTOK / 256), 256, 0, stream>>>(
        scaledb, MAXP_, WTin + (size_t)p * D_ * MAXP_, MAXP_, b_in + p * D_,
        xin + (size_t)p * NTOK * D_, nullptr, (long)D_, 1L, NTOK, D_, psizes[p], 0);
  }
  combine_x<<<NTOK * D_ / 256, 256, 0, stream>>>(xin, patch_sz, pred_mask, mask_emb, x);

  // ---- transformer layers ----
  for (int l = 0; l < L_; ++l) {
    rmsnorm_768<<<NTOK, 256, 0, stream>>>(x, norm1_w + l * D_, hb);
    gemm_wmma<<<dim3(D_ / 64, NTOK / 256), 256, 0, stream>>>(
        hb, D_, WTq + (size_t)l * D_ * D_, D_, nullptr, qf, nullptr,
        (long)D_, 1L, NTOK, D_, D_, 0);
    gemm_wmma<<<dim3(D_ / 64, NTOK / 256), 256, 0, stream>>>(
        hb, D_, WTk + (size_t)l * D_ * D_, D_, nullptr, kf, nullptr,
        (long)D_, 1L, NTOK, D_, D_, 0);
    for (int b = 0; b < B_; ++b) {  // V stored transposed: [H,HD,S] per batch
      gemm_wmma<<<dim3(D_ / 64, S_ / 256), 256, 0, stream>>>(
          hb + (size_t)b * S_ * D_, D_, WTv + (size_t)l * D_ * D_, D_, nullptr,
          vt + (size_t)b * D_ * S_, nullptr, 1L, (long)S_, S_, D_, D_, 2);
    }
    qknorm_rope<<<NTOK * H_ / 8, 256, 0, stream>>>(qf, qb2, q_norm_w + l * HD_, time_id);
    qknorm_rope<<<NTOK * H_ / 8, 256, 0, stream>>>(kf, kb2, k_norm_w + l * HD_, time_id);
    attn_fused<<<B_ * H_ * (S_ / 16) / 8, 256, 0, stream>>>(
        qb2, kb2, vt, ob, sample_id, variate_id, var_bias, l);
    gemm_wmma<<<dim3(D_ / 64, NTOK / 256), 256, 0, stream>>>(   // x += o @ Wo
        ob, D_, WTo + (size_t)l * D_ * D_, D_, nullptr, x, nullptr,
        (long)D_, 1L, NTOK, D_, D_, 1);
    rmsnorm_768<<<NTOK, 256, 0, stream>>>(x, norm2_w + l * D_, hb);
    gemm_wmma<<<dim3(DFF_ / 64, NTOK / 256), 256, 0, stream>>>(  // g1 = h @ W1
        hb, D_, WT1 + (size_t)l * DFF_ * D_, D_, nullptr, g1, nullptr,
        (long)DFF_, 1L, NTOK, DFF_, D_, 0);
    gemm_wmma<<<dim3(DFF_ / 64, NTOK / 256), 256, 0, stream>>>(  // gl = silu(g1)*(h@W3)
        hb, D_, WT3 + (size_t)l * DFF_ * D_, D_, nullptr, gl, g1,
        (long)DFF_, 1L, NTOK, DFF_, D_, 3);
    gemm_wmma<<<dim3(D_ / 64, NTOK / 256), 256, 0, stream>>>(    // x += gl @ W2
        gl, DFF_, WT2 + (size_t)l * D_ * DFF_, DFF_, nullptr, x, nullptr,
        (long)D_, 1L, NTOK, D_, DFF_, 1);
  }

  // ---- output heads ----
  rmsnorm_768<<<NTOK, 256, 0, stream>>>(x, fnorm_w, hb);
  for (int p = 0; p < 3; ++p) {
    gemm_wmma<<<dim3(MAXP_ / 64, NTOK / 256), 256, 0, stream>>>(
        hb, D_, WTl + (size_t)p * MAXP_ * D_, D_, b_loc + p * MAXP_,
        outp + (size_t)p * NTOK * MAXP_, nullptr, (long)MAXP_, 1L, NTOK, MAXP_, D_, 0);
    gemm_wmma<<<dim3(MAXP_ / 64, NTOK / 256), 256, 0, stream>>>(
        hb, D_, WTs + (size_t)p * MAXP_ * D_, D_, b_scale + p * MAXP_,
        outp + (size_t)(3 + p) * NTOK * MAXP_, nullptr, (long)MAXP_, 1L, NTOK, MAXP_, D_, 0);
  }
  final_out<<<NTOK * MAXP_ / 256, 256, 0, stream>>>(outp, patch_sz, locb, scalev,
                                                    (float*)d_out);
}
